// CRF_15753940042185
// MI455X (gfx1250) — matrix operand order repaired and checked by reading, not compiled, not run on hardware
//
#include <hip/hip_runtime.h>
#include <math.h>

typedef __attribute__((ext_vector_type(16))) _Float16 v16h;
typedef __attribute__((ext_vector_type(8)))  float    v8f;

#define S_LEN       8192
#define TSZ         2048
#define NEG_VAL     (-10000.0f)
#define START_I     0
#define END_I       1
#define NWG         128
#define ROWS_PER_WG 16            // TSZ / NWG
#define NTHREADS    256
#define NWAVES      8             // wave32
#define K_PER_WAVE  (TSZ / NWAVES)      // 256
#define KT_PER_WAVE (K_PER_WAVE / 32)   // 8 k-tiles of 32

// ---- dynamic LDS partition (bytes) ----
#define SE_HALFS   (ROWS_PER_WG * TSZ)          // 32768 halfs = 65536 B (E slice, swizzled)
#define OFF_SV     (SE_HALFS * 2)               // 65536
#define OFF_PART   (OFF_SV + TSZ * 2)           // 69632  (NWAVES*16 floats)
#define OFF_RED    (OFF_PART + NWAVES * 16 * 4) // 70144  (256 floats)
#define OFF_CMAX   (OFF_RED + NTHREADS * 4)     // 71168  (16 floats)
#define SMEM_BYTES (OFF_CMAX + ROWS_PER_WG * 4) // 71232

extern __shared__ __align__(32) char smem_raw[];

__device__ inline void grid_barrier(unsigned* counter, unsigned target) {
    __syncthreads();
    if (threadIdx.x == 0) {
        __threadfence();  // make this WG's global writes visible device-wide
        __hip_atomic_fetch_add(counter, 1u, __ATOMIC_ACQ_REL, __HIP_MEMORY_SCOPE_AGENT);
        while (__hip_atomic_load(counter, __ATOMIC_ACQUIRE, __HIP_MEMORY_SCOPE_AGENT) < target) {
            __builtin_amdgcn_s_sleep(1);
        }
    }
    __syncthreads();
}

__global__ void crf_init(float* fbuf0, unsigned* counter) {
    int i = blockIdx.x * blockDim.x + threadIdx.x;
    if (i == 0) *counter = 0u;
    if (i < TSZ) fbuf0[i] = (i == START_I) ? 0.0f : NEG_VAL;
}

__global__ void crf_persistent(const float* __restrict__ h,
                               const float* __restrict__ trans,
                               float* __restrict__ out,
                               float* fbuf0, float* fbuf1,
                               unsigned* counter) {
    _Float16* sE    = (_Float16*)(smem_raw);            // [16][2048] swizzled A tiles
    _Float16* sV    = (_Float16*)(smem_raw + OFF_SV);   // exp(f - m) vector, f16
    float*    sPart = (float*)   (smem_raw + OFF_PART); // [NWAVES][16] partial sums
    float*    sRed  = (float*)   (smem_raw + OFF_RED);  // [256] reduction scratch
    float*    sCmax = (float*)   (smem_raw + OFF_CMAX); // [16] per-row max of T

    const int tid  = threadIdx.x;
    const int wid  = tid >> 5;
    const int lane = tid & 31;
    const int row0 = blockIdx.x * ROWS_PER_WG;

    // ---------- one-time init: per-row max of transitions ----------
    {
        int r = tid >> 4, c = tid & 15;
        const float* trow = trans + (size_t)(row0 + r) * TSZ;
        float mm = -INFINITY;
        for (int k = c; k < TSZ; k += 16) mm = fmaxf(mm, trow[k]);
        sRed[tid] = mm;
        __syncthreads();
        if (c == 0) {
            float mx = sRed[r * 16];
            #pragma unroll
            for (int i = 1; i < 16; ++i) mx = fmaxf(mx, sRed[r * 16 + i]);
            sCmax[r] = mx;
        }
        __syncthreads();
    }

    // ---------- one-time init: E = exp(T - c_i), swizzled into WMMA A layout ----------
    // A 16x32 tile as packed here (contiguous k; B uses the same): lane<16 holds row, K{0..15}; lane>=16 holds K{16..31}
    for (int e = tid; e < ROWS_PER_WG * TSZ; e += NTHREADS) {
        int r = e >> 11;
        int k = e & (TSZ - 1);
        float val = __expf(trans[(size_t)(row0 + r) * TSZ + k] - sCmax[r]);
        int tile = k >> 5;
        int kk   = k & 31;
        int half = kk >> 4;
        int ln   = r + (half << 4);
        int slot = kk & 15;
        sE[tile * 512 + ln * 16 + slot] = (_Float16)val;
    }
    __syncthreads();

    const _Float16* aBase = sE + (wid * KT_PER_WAVE) * 512 + lane * 16;
    const _Float16* bBase0 = sV + wid * K_PER_WAVE + ((lane & 16) ? 16 : 0);

    // ---------- sequential scan ----------
    for (int s = 0; s < S_LEN; ++s) {
        grid_barrier(counter, (unsigned)(s + 1) * NWG);

        const float* fc = (s & 1) ? fbuf1 : fbuf0;
        float*       fn = (s & 1) ? fbuf0 : fbuf1;

        // global max of f (identical in every WG)
        float fvals[8];
        float lmax = -INFINITY;
        #pragma unroll
        for (int k = 0; k < 8; ++k) {
            float x = fc[tid + k * NTHREADS];
            fvals[k] = x;
            lmax = fmaxf(lmax, x);
        }
        sRed[tid] = lmax;
        __syncthreads();
        #pragma unroll
        for (int off = 128; off > 0; off >>= 1) {
            if (tid < off) sRed[tid] = fmaxf(sRed[tid], sRed[tid + off]);
            __syncthreads();
        }
        float m = sRed[0];

        // v = exp(f - m) as f16
        #pragma unroll
        for (int k = 0; k < 8; ++k)
            sV[tid + k * NTHREADS] = (_Float16)__expf(fvals[k] - m);
        __syncthreads();

        // GEMV slice via chained WMMA: acc += A(16x32) * v-replicated(32x16)
        v8f acc = {};
        #pragma unroll
        for (int t = 0; t < KT_PER_WAVE; ++t) {
            v16h a = *(const v16h*)(aBase + t * 512);
            v16h b = *(const v16h*)(bBase0 + t * 32);
            acc = __builtin_amdgcn_wmma_f32_16x16x32_f16(
                false, a, false, b, (short)0, acc, false, false);
        }

        // C layout: lane0 col0 rows0-7 in acc[0..7]; lane16 col0 rows8-15
        if (lane == 0 || lane == 16) {
            int rbase = (lane >> 4) << 3;
            #pragma unroll
            for (int r = 0; r < 8; ++r) sPart[wid * 16 + rbase + r] = acc[r];
        }
        __syncthreads();

        if (tid < ROWS_PER_WG) {
            float ssum = 0.0f;
            #pragma unroll
            for (int w = 0; w < NWAVES; ++w) ssum += sPart[w * 16 + tid];
            int row = row0 + tid;
            fn[row] = __logf(ssum) + sCmax[tid] + m + h[(size_t)s * TSZ + row];
        }
        // visibility handled by next grid_barrier
    }

    grid_barrier(counter, (unsigned)(S_LEN + 1) * NWG);

    // ---------- final logsumexp(f + T[END,:]) on WG 0 ----------
    if (blockIdx.x == 0) {
        const float* ff = (S_LEN & 1) ? fbuf1 : fbuf0;
        const float* te = trans + (size_t)END_I * TSZ;
        float tv[8];
        float lmax = -INFINITY;
        #pragma unroll
        for (int k = 0; k < 8; ++k) {
            float x = ff[tid + k * NTHREADS] + te[tid + k * NTHREADS];
            tv[k] = x;
            lmax = fmaxf(lmax, x);
        }
        sRed[tid] = lmax;
        __syncthreads();
        #pragma unroll
        for (int off = 128; off > 0; off >>= 1) {
            if (tid < off) sRed[tid] = fmaxf(sRed[tid], sRed[tid + off]);
            __syncthreads();
        }
        float M = sRed[0];
        __syncthreads();
        float ls = 0.0f;
        #pragma unroll
        for (int k = 0; k < 8; ++k) ls += __expf(tv[k] - M);
        sRed[tid] = ls;
        __syncthreads();
        #pragma unroll
        for (int off = 128; off > 0; off >>= 1) {
            if (tid < off) sRed[tid] += sRed[tid + off];
            __syncthreads();
        }
        if (tid == 0) out[0] = __logf(sRed[0]) + M;
    }
}

extern "C" void kernel_launch(void* const* d_in, const int* in_sizes, int n_in,
                              void* d_out, int out_size, void* d_ws, size_t ws_size,
                              hipStream_t stream) {
    const float* h     = (const float*)d_in[0];   // [8192, 2048]
    const float* trans = (const float*)d_in[1];   // [2048, 2048]
    float* out = (float*)d_out;

    unsigned* counter = (unsigned*)d_ws;
    float* fbuf0 = (float*)((char*)d_ws + 256);
    float* fbuf1 = fbuf0 + TSZ;

    (void)hipFuncSetAttribute((const void*)crf_persistent,
                              hipFuncAttributeMaxDynamicSharedMemorySize,
                              (int)SMEM_BYTES);

    crf_init<<<(TSZ + 255) / 256, 256, 0, stream>>>(fbuf0, counter);
    crf_persistent<<<NWG, NTHREADS, SMEM_BYTES, stream>>>(h, trans, out,
                                                          fbuf0, fbuf1, counter);
}
